// SP_Layer_61100204753306
// MI455X (gfx1250) — 8-bit matrix operand order repaired and checked by reading, compiled, not run on hardware
//
#include <hip/hip_runtime.h>

typedef __attribute__((ext_vector_type(8))) int v8i;

#define SP_SIZE   8192
#define SP_INPUT  8192
#define SP_NNEI   2048
#define SP_KTH    26
#define SP_KSPLIT 8
#define SP_KLEN   (SP_INPUT / SP_KSPLIT)
#define SP_KITERS (SP_KLEN / 64)

__global__ void sp_zero_ws(int* __restrict__ ws) {
    int i = blockIdx.x * blockDim.x + threadIdx.x;
    if (i < SP_SIZE) ws[i] = 0;
}

__global__ void __launch_bounds__(256) sp_overlap_wmma(const int*   __restrict__ input,
                                                       const float* __restrict__ perms,
                                                       int*         __restrict__ ws) {
    __shared__ unsigned int s_inp[SP_INPUT / 4];

    const int tid = threadIdx.x;
    for (int i = tid; i < SP_INPUT / 4; i += 256) {
        const int4 w = reinterpret_cast<const int4*>(input)[i];
        unsigned b = (w.x > 0 ? 0x1u : 0u) | (w.y > 0 ? 0x100u : 0u)
                   | (w.z > 0 ? 0x10000u : 0u) | (w.w > 0 ? 0x1000000u : 0u);
        s_inp[i] = b;
    }
    __syncthreads();

    const int wave = blockIdx.x * 8 + (tid >> 5);
    const int lane = tid & 31;
    const int rb   = wave >> 3;
    const int kb   = wave & 7;
    const int row  = rb * 16 + (lane & 15);
    const int hi16 = (lane >= 16);
    const int koff = hi16 ? 16 : 0;

    const float* p = perms + (size_t)row * SP_SIZE + kb * SP_KLEN + koff;

    const int4* bptr = reinterpret_cast<const int4*>(s_inp)
                     + kb * (SP_KLEN / 16) + hi16;

    v8i acc = {};
    for (int kc = 0; kc < SP_KITERS; ++kc) {
        __builtin_prefetch(p + 64, 0, 1);

        const float4 p0 = *reinterpret_cast<const float4*>(p +  0);
        const float4 p1 = *reinterpret_cast<const float4*>(p +  4);
        const float4 p2 = *reinterpret_cast<const float4*>(p +  8);
        const float4 p3 = *reinterpret_cast<const float4*>(p + 12);
        const float4 p4 = *reinterpret_cast<const float4*>(p + 32);
        const float4 p5 = *reinterpret_cast<const float4*>(p + 36);
        const float4 p6 = *reinterpret_cast<const float4*>(p + 40);
        const float4 p7 = *reinterpret_cast<const float4*>(p + 44);
        p += 64;

        const int4 b0 = bptr[kc * 4];
        const int4 b1 = bptr[kc * 4 + 2];

#define SP_PACK(q) (int)((q.x > 0.6f ? 0x1u : 0u) | (q.y > 0.6f ? 0x100u : 0u) \
                   | (q.z > 0.6f ? 0x10000u : 0u) | (q.w > 0.6f ? 0x1000000u : 0u))
        v8i A, B;
        A[0] = SP_PACK(p0); A[1] = SP_PACK(p1); A[2] = SP_PACK(p2); A[3] = SP_PACK(p3);
        A[4] = SP_PACK(p4); A[5] = SP_PACK(p5); A[6] = SP_PACK(p6); A[7] = SP_PACK(p7);
#undef SP_PACK
        B[0] = b0.x; B[1] = b0.y; B[2] = b0.z; B[3] = b0.w;
        B[4] = b1.x; B[5] = b1.y; B[6] = b1.z; B[7] = b1.w;

        acc = __builtin_amdgcn_wmma_i32_16x16x64_iu8(false, A, false, B, acc, false, false);
    }

    if (lane == 0 || lane == 16) {
        const int rbase = rb * 16 + (hi16 ? 8 : 0);
#pragma unroll
        for (int r = 0; r < 8; ++r) atomicAdd(ws + rbase + r, acc[r]);
    }
}

__global__ void __launch_bounds__(256) sp_select(const int* __restrict__ ws,
                                                 const int* __restrict__ nidx,
                                                 float*     __restrict__ out) {
    __shared__ unsigned short s_ov[SP_SIZE];
    const int tid = threadIdx.x;
    for (int i = tid; i < SP_SIZE; i += 256) s_ov[i] = (unsigned short)ws[i];
    __syncthreads();

    const int lane   = tid & 31;
    const int waveg  = blockIdx.x * 8 + (tid >> 5);
    const int nwaves = gridDim.x * 8;

    for (int row = waveg; row < SP_SIZE; row += nwaves) {
        const int* nrow = nidx + (size_t)row * SP_NNEI + lane;
        int vals[SP_NNEI / 32];
#pragma unroll
        for (int t = 0; t < SP_NNEI / 32; ++t)
            vals[t] = (int)s_ov[nrow[t * 32]];

        int lo = 0, hi = SP_SIZE + 1;
        while (hi - lo > 1) {
            const int mid = (lo + hi) >> 1;
            int c = 0;
#pragma unroll
            for (int t = 0; t < SP_NNEI / 32; ++t) c += (vals[t] >= mid) ? 1 : 0;
#pragma unroll
            for (int off = 16; off > 0; off >>= 1) c += __shfl_xor(c, off, 32);
            if (c >= SP_KTH) lo = mid; else hi = mid;
        }
        if (lane == 0) out[row] = ((int)s_ov[row] > lo) ? 1.0f : 0.0f;
    }
}

extern "C" void kernel_launch(void* const* d_in, const int* in_sizes, int n_in,
                              void* d_out, int out_size, void* d_ws, size_t ws_size,
                              hipStream_t stream) {
    const int*   input = (const int*)d_in[0];
    const float* perms = (const float*)d_in[1];
    const int*   nidx  = (const int*)d_in[2];
    float* out = (float*)d_out;
    int*   ws  = (int*)d_ws;

    sp_zero_ws<<<SP_SIZE / 256, 256, 0, stream>>>(ws);
    sp_overlap_wmma<<<512, 256, 0, stream>>>(input, perms, ws);
    sp_select<<<128, 256, 0, stream>>>(ws, nidx, out);
}
